// GraphConvLayer_55714315764268
// MI455X (gfx1250) — matrix operand order repaired and checked by reading, not compiled, not run on hardware
//
#include <hip/hip_runtime.h>
#include <hip/hip_bf16.h>
#include <cstddef>

// Problem constants
#define BB   64
#define NN   784
#define FF   64
#define KP   800          // K dimension padded to multiple of 32 (784 -> 800)
#define COLS 4160         // 64 batches * 64 feats + 64 denominator columns
#define CSTRIPS 65        // COLS / 64
#define ITILES  49        // NN / 16
#define NWAVES  (ITILES * CSTRIPS)

typedef _Float16 v16h __attribute__((ext_vector_type(16)));
typedef _Float16 v8h  __attribute__((ext_vector_type(8)));
typedef float    v8f  __attribute__((ext_vector_type(8)));

// ---------------------------------------------------------------------------
// Kernel 1: projections (fp32 VALU — only 0.8 GFLOP, keep exact)
//   self[b,j,:] = X W_self^T + b_self ; nb[b,j,:] = X W_nb^T + b_nb
//   dst[b,j]    = X . a_dst
// One block per (b,j) row; 128 threads: t<64 -> self feat t, t>=64 -> nb feat.
// ---------------------------------------------------------------------------
__global__ __launch_bounds__(128) void gat_project(
    const float* __restrict__ X,
    const float* __restrict__ Wself, const float* __restrict__ bself,
    const float* __restrict__ Wnb,   const float* __restrict__ bnb,
    const float* __restrict__ adst,
    float* __restrict__ selfF, float* __restrict__ nbF, float* __restrict__ dstF)
{
    __shared__ float xs[FF];
    const int row = blockIdx.x;              // b*NN + j
    const int t   = threadIdx.x;
    const float* xp = X + (size_t)row * FF;
    if (t < FF) xs[t] = xp[t];
    __syncthreads();

    const int f = t & 63;
    const float* W = (t < FF) ? (Wself + f * FF) : (Wnb + f * FF);
    float acc = (t < FF) ? bself[f] : bnb[f];
#pragma unroll 8
    for (int k = 0; k < FF; ++k) acc += xs[k] * W[k];
    if (t < FF) selfF[(size_t)row * FF + f] = acc;
    else        nbF  [(size_t)row * FF + f] = acc;

    if (t == 0) {
        float d = 0.f;
#pragma unroll 8
        for (int k = 0; k < FF; ++k) d += xs[k] * adst[k];
        dstF[row] = d;
    }
}

// ---------------------------------------------------------------------------
// Kernel 2: adjacency -> f16 A-operand, zero-padded to KP columns
// ---------------------------------------------------------------------------
__global__ __launch_bounds__(256) void gat_maskcvt(
    const int* __restrict__ adj, _Float16* __restrict__ Amask)
{
    int idx = blockIdx.x * 256 + threadIdx.x;
    if (idx >= NN * KP) return;
    int i = idx / KP, k = idx % KP;
    Amask[idx] = (k < NN && adj[i * NN + k] != 0) ? (_Float16)1.0f : (_Float16)0.0f;
}

// ---------------------------------------------------------------------------
// Kernel 3: per-batch softmax prep.  M_b = max_j dst; e_j = exp(dst-M).
// Build G (K-contiguous f16 B-operand):
//   G[b*64+f][j] = e_j * nb[b,j,f]      (numerator columns)
//   G[4096+b][j] = e_j                  (denominator column)
// j in [784,800) zero-padded.  One block per batch.
// ---------------------------------------------------------------------------
__global__ __launch_bounds__(256) void gat_softprep(
    const float* __restrict__ dstF, const float* __restrict__ nbF,
    _Float16* __restrict__ G)
{
    __shared__ float sE[NN];
    __shared__ float sred[256];
    const int b = blockIdx.x;
    const int t = threadIdx.x;

    float lmax = -1e30f;
    for (int j = t; j < NN; j += 256) {
        float d = dstF[b * NN + j];
        sE[j] = d;
        lmax = fmaxf(lmax, d);
    }
    sred[t] = lmax;
    __syncthreads();
    for (int s = 128; s > 0; s >>= 1) {
        if (t < s) sred[t] = fmaxf(sred[t], sred[t + s]);
        __syncthreads();
    }
    const float M = sred[0];
    __syncthreads();
    for (int j = t; j < NN; j += 256) sE[j] = expf(sE[j] - M);
    __syncthreads();

    const float* nbB = nbF + (size_t)b * NN * FF;
    for (int idx = t; idx < FF * KP; idx += 256) {
        int f = idx / KP, j = idx % KP;
        _Float16 v = (j < NN) ? (_Float16)(sE[j] * nbB[(size_t)j * FF + f])
                              : (_Float16)0.0f;
        G[(size_t)(b * FF + f) * KP + j] = v;
    }
    for (int j = t; j < KP; j += 256)
        G[(size_t)(BB * FF + b) * KP + j] =
            (j < NN) ? (_Float16)sE[j] : (_Float16)0.0f;
}

// ---------------------------------------------------------------------------
// Kernel 4: the big WMMA GEMM.  C[784 x 4160] = Amask[784 x 800] * G^T
// One wave per 16x64 output strip; A fragment reused across 4 B fragments.
// A-layout (16-bit 16x32): lane m=lane&15, halves 0..7 -> K = 8*(lane>>4)+0..7,
// halves 8..15 -> +16: two b128 loads.  B: column N=lane&15, K contiguous.
// ---------------------------------------------------------------------------
__global__ __launch_bounds__(128) void gat_wmma_gemm(
    const _Float16* __restrict__ Amask, const _Float16* __restrict__ G,
    float* __restrict__ C)
{
    const int wave = blockIdx.x * (blockDim.x >> 5) + (threadIdx.x >> 5);
    if (wave >= NWAVES) return;                 // whole-wave uniform exit
    const int lane  = threadIdx.x & 31;
    const int itile = wave / CSTRIPS;
    const int strip = wave % CSTRIPS;
    const int i0 = itile * 16;
    const int c0 = strip * 64;
    const int m  = lane & 15;
    const int hi = lane >> 4;

    const _Float16* aRow = Amask + (size_t)(i0 + m) * KP + hi * 8;

    v8f acc[4] = {v8f{}, v8f{}, v8f{}, v8f{}};

    for (int k0 = 0; k0 < KP; k0 += 32) {
        v8h alo = *(const v8h*)(aRow + k0);        // K = klane+0..7
        v8h ahi = *(const v8h*)(aRow + k0 + 16);   // K = klane+16..23
        v16h a;
#pragma unroll
        for (int h = 0; h < 8; ++h) { a[h] = alo[h]; a[h + 8] = ahi[h]; }

#pragma unroll
        for (int tix = 0; tix < 4; ++tix) {
            const _Float16* bptr =
                G + (size_t)(c0 + 16 * tix + m) * KP + (hi << 3) + k0;
            v8h blo = *(const v8h*)(bptr);         // K = kb+0..7
            v8h bhi = *(const v8h*)(bptr + 16);    // K = kb+16..23
            v16h bf;
#pragma unroll
            for (int h = 0; h < 8; ++h) { bf[h] = blo[h]; bf[h + 8] = bhi[h]; }
            acc[tix] = __builtin_amdgcn_wmma_f32_16x16x32_f16(
                false, a, false, bf, (short)0, acc[tix], false, false);
        }
    }

#pragma unroll
    for (int tix = 0; tix < 4; ++tix) {
        const int col = c0 + 16 * tix + m;
#pragma unroll
        for (int r = 0; r < 8; ++r) {
            const int rowi = i0 + r + (hi << 3);
            C[(size_t)rowi * COLS + col] = acc[tix][r];
        }
    }
}

// ---------------------------------------------------------------------------
// Kernel 5: finalize. combined = self + num/den ; LayerNorm over F ; ReLU.
// One wave32 per (b,i); each lane owns features lane and lane+32.
// ---------------------------------------------------------------------------
__global__ __launch_bounds__(256) void gat_finalize(
    const float* __restrict__ selfF, const float* __restrict__ C,
    const float* __restrict__ gamma, const float* __restrict__ beta,
    float* __restrict__ out)
{
    const int w = blockIdx.x * 8 + (threadIdx.x >> 5);
    if (w >= BB * NN) return;
    const int lane = threadIdx.x & 31;
    const int b = w / NN;
    const int i = w % NN;

    const float* crow = C + (size_t)i * COLS;
    const float den = crow[BB * FF + b];
    const float inv = (den > 0.f) ? (1.f / den) : 0.f;   // isolated node -> agg 0

    const int f0 = lane, f1 = lane + 32;
    const size_t base = (size_t)w * FF;
    float v0 = selfF[base + f0] + crow[b * FF + f0] * inv;
    float v1 = selfF[base + f1] + crow[b * FF + f1] * inv;

    float s = v0 + v1, q = v0 * v0 + v1 * v1;
    for (int o = 16; o; o >>= 1) {
        s += __shfl_xor(s, o, 32);
        q += __shfl_xor(q, o, 32);
    }
    const float mu  = s * (1.f / FF);
    const float var = q * (1.f / FF) - mu * mu;
    const float rs  = rsqrtf(var + 1e-5f);
    out[base + f0] = fmaxf(0.f, (v0 - mu) * rs * gamma[f0] + beta[f0]);
    out[base + f1] = fmaxf(0.f, (v1 - mu) * rs * gamma[f1] + beta[f1]);
}

// ---------------------------------------------------------------------------
// Host launch
// ---------------------------------------------------------------------------
extern "C" void kernel_launch(void* const* d_in, const int* in_sizes, int n_in,
                              void* d_out, int out_size, void* d_ws, size_t ws_size,
                              hipStream_t stream)
{
    const float* X      = (const float*)d_in[0];
    const int*   adj    = (const int*)  d_in[1];
    const float* Wself  = (const float*)d_in[2];
    const float* bself  = (const float*)d_in[3];
    const float* Wnb    = (const float*)d_in[4];
    const float* bnb    = (const float*)d_in[5];
    // d_in[6] = a_src, d_in[8] = b_att: cancel in the row softmax, unused.
    const float* adst   = (const float*)d_in[7];
    const float* gamma  = (const float*)d_in[9];
    const float* beta   = (const float*)d_in[10];
    float* out = (float*)d_out;

    // Workspace layout (all offsets 256B-aligned); total ~46.85 MB
    char* ws = (char*)d_ws;
    size_t off = 0;
    float*    selfF = (float*)(ws + off);    off += (size_t)BB * NN * FF * 4;   // 12.85 MB
    float*    nbF   = (float*)(ws + off);    off += (size_t)BB * NN * FF * 4;   // 12.85 MB
    float*    dstF  = (float*)(ws + off);    off += (size_t)BB * NN * 4;        // 0.20 MB
    _Float16* Amask = (_Float16*)(ws + off); off += (size_t)NN * KP * 2;        // 1.25 MB
    _Float16* G     = (_Float16*)(ws + off); off += (size_t)COLS * KP * 2;      // 6.66 MB
    float*    Cbuf  = (float*)(ws + off);    off += (size_t)NN * COLS * 4;      // 13.05 MB

    gat_project<<<BB * NN, 128, 0, stream>>>(X, Wself, bself, Wnb, bnb, adst,
                                             selfF, nbF, dstF);
    gat_maskcvt<<<(NN * KP + 255) / 256, 256, 0, stream>>>(adj, Amask);
    gat_softprep<<<BB, 256, 0, stream>>>(dstF, nbF, G);
    gat_wmma_gemm<<<(NWAVES + 3) / 4, 128, 0, stream>>>(Amask, G, Cbuf);
    gat_finalize<<<(BB * NN + 7) / 8, 256, 0, stream>>>(selfF, Cbuf, gamma, beta, out);
    (void)in_sizes; (void)n_in; (void)out_size; (void)ws_size;
}
